// RNNLangModel_58875411693629
// MI455X (gfx1250) — matrix operand order repaired and checked by reading, not compiled, not run on hardware
//
#include <hip/hip_runtime.h>

// ---------------------------------------------------------------------------
// LSTM language model for MI455X (gfx1250, wave32, WMMA bf16).
//   tokens = embed[ids]                      [B,S,E]
//   scan:  gates = [x_t|h] @ W{f,i,c,o} + b ; c = c*f + g*i ; h = c*o
//   out   = h @ Wout + bout                  [B,V] f32
// Strategy: pre-pack everything into WMMA fragment layouts in bf16, run one
// fused kernel per timestep (weights stay L2-resident, A staged in LDS),
// ping-pong h in fragment layout, final projection with the same tiles.
// Inner GEMM loops are software-pipelined (double-buffered A/B registers) so
// ds_load/global_load latency hides under the 4-WMMA issue shadow.
// ---------------------------------------------------------------------------

#define VOCAB  32000
#define EMBED  512
#define HIDDEN 1024
#define BATCH  64
#define SEQ    512
#define DCAT   (EMBED + HIDDEN)   // 1536
#define KT_X   (EMBED / 32)       // 16 K-tiles from x
#define KT_H   (HIDDEN / 32)      // 32 K-tiles from h
#define KT_ALL (DCAT / 32)        // 48
#define MT     (BATCH / 16)       // 4 M-tiles
#define NT_H   (HIDDEN / 16)      // 64 N-tiles (hidden)
#define NT_V   (VOCAB / 16)       // 2000 N-tiles (vocab)
#define FRAG   512                // elements per 16x16x32 fragment (32 lanes x 16)

typedef __bf16 v16bf __attribute__((ext_vector_type(16)));
typedef float  v8f   __attribute__((ext_vector_type(8)));

__device__ __forceinline__ v8f wmma_bf16(v16bf a, v16bf b, v8f c) {
  // D = A(16x32 bf16) * B(32x16 bf16) + C(16x16 f32)
  return __builtin_amdgcn_wmma_f32_16x16x32_bf16(
      /*neg_a=*/false, a, /*neg_b=*/false, b,
      /*c_mod=*/(short)0, c, /*reuse_a=*/false, /*reuse_b=*/false);
}

// A-matrix (16x32, 16-bit) lane/slot -> K mapping (ISA 7.12.2):
//   lane = m + 16*g ; slot s (=2*v+h) covers K per:
//   k = (v<4 ? 2v+h+8g : 16+2(v-4)+h+8g)
__device__ __forceinline__ int a_slot_to_k(int slot, int g) {
  int v = slot >> 1, hb = slot & 1;
  return (v < 4) ? (2 * v + hb + 8 * g) : (16 + 2 * (v - 4) + hb + 8 * g);
}

// ------------------------- packing kernels ---------------------------------

// Zero c state and the initial h fragment buffer (both BATCH*HIDDEN elems).
__global__ void zero_state(float* __restrict__ c, __bf16* __restrict__ h0) {
  int t = blockIdx.x * 256 + threadIdx.x;   // 65536 threads
  c[t]  = 0.0f;
  h0[t] = (__bf16)0.0f;
}

// Embedding gather + bf16 conversion into A-fragment layout, all timesteps.
// One thread per (s, mt, kt, lane): writes 16 contiguous bf16 (32B).
__global__ void pack_x(const int* __restrict__ ids,
                       const float* __restrict__ embed,
                       __bf16* __restrict__ xp) {
  int tid  = blockIdx.x * 256 + threadIdx.x;      // 512*4*16*32 = 1,048,576
  int lane = tid & 31;
  int kt   = (tid >> 5) & 15;
  int mt   = (tid >> 9) & 3;
  int s    = tid >> 11;
  int m = lane & 15, gA = lane >> 4;
  int b = mt * 16 + m;
  int id = ids[b * SEQ + s];
  const float* row = embed + (size_t)id * EMBED;
  v16bf frag;
#pragma unroll
  for (int slot = 0; slot < 16; ++slot) {
    int e = kt * 32 + a_slot_to_k(slot, gA);
    frag[slot] = (__bf16)row[e];
  }
  *(v16bf*)(xp + ((size_t)((s * MT + mt) * KT_X + kt)) * FRAG + lane * 16) = frag;
}

// Gate weights [DCAT,HIDDEN] f32 -> bf16 B-fragment layout [g][nt][kt][512].
// B-matrix (32x16, 16-bit): n = lane&15, k = (lane>>4)*8 + (slot&7) + (slot>>3)*16.
__global__ void pack_w(const float* __restrict__ Wf, const float* __restrict__ Wi,
                       const float* __restrict__ Wc, const float* __restrict__ Wo,
                       __bf16* __restrict__ wp) {
  int tid  = blockIdx.x * 256 + threadIdx.x;      // 4*64*48*32 = 393,216
  int lane = tid & 31;
  int fid  = tid >> 5;                            // 0..12287
  int kt = fid % KT_ALL;
  int nt = (fid / KT_ALL) % NT_H;
  int g  = fid / (KT_ALL * NT_H);
  const float* W = (g == 0) ? Wf : (g == 1) ? Wi : (g == 2) ? Wc : Wo;
  int n = lane & 15;
  int kbase = (lane >> 4) * 8;
  v16bf frag;
#pragma unroll
  for (int slot = 0; slot < 16; ++slot) {
    frag[slot] = (__bf16)W[(size_t)(kt * 32 + kbase + (slot & 7) + ((slot >> 3) << 4)) * HIDDEN + nt * 16 + n];
  }
  *(v16bf*)(wp + (size_t)fid * FRAG + lane * 16) = frag;
}

// Wout [HIDDEN,VOCAB] f32 -> bf16 B-fragment layout [nt][kt][512].
__global__ void pack_wout(const float* __restrict__ Wout, __bf16* __restrict__ wp) {
  int tid  = blockIdx.x * 256 + threadIdx.x;      // 2000*32*32 = 2,048,000
  int lane = tid & 31;
  int fid  = tid >> 5;                            // 0..63999
  int kt = fid & 31;
  int nt = fid >> 5;
  int n = lane & 15;
  int kbase = (lane >> 4) * 8;
  v16bf frag;
#pragma unroll
  for (int slot = 0; slot < 16; ++slot) {
    frag[slot] = (__bf16)Wout[(size_t)(kt * 32 + kbase + (slot & 7) + ((slot >> 3) << 4)) * VOCAB + nt * 16 + n];
  }
  *(v16bf*)(wp + (size_t)fid * FRAG + lane * 16) = frag;
}

// ------------------------- recurrent step ----------------------------------

__device__ __forceinline__ void activate_store(float* __restrict__ Glds, v8f acc,
                                               int g, int mt, int lane, float bias) {
  const int colj  = lane & 15;
  const int rbase = 8 * (lane >> 4);
#pragma unroll
  for (int r = 0; r < 8; ++r) {
    float x = acc[r] + bias;
    float y;
    if (g == 2) {                     // candidate gate: tanh (safe at +-inf)
      float e = __expf(2.0f * x);
      y = 1.0f - 2.0f / (e + 1.0f);
    } else {                          // f, i, o: sigmoid
      y = 1.0f / (1.0f + __expf(-x));
    }
    Glds[g * (BATCH * 16) + (mt * 16 + rbase + r) * 16 + colj] = y;
  }
}

// One timestep. grid = 64 (one per 16 hidden columns), block = 128 (4 waves,
// one per gate). Dynamic LDS: 4*48*512 bf16 A-stage (192KB) + 4*64*16 f32
// gate buffer (16KB) -> 208KB of the WGP's 320KB.
__global__ void __launch_bounds__(128) lstm_step(
    const __bf16* __restrict__ xp_s,   // A-frags of x_t: [mt][kt<16][512]
    const __bf16* __restrict__ wp,     // gate weight B-frags
    const __bf16* __restrict__ hin,    // A-frags of h_{t-1}: [mt][kt<32][512]
    __bf16* __restrict__ hout,         // A-frags of h_t (scatter-written)
    float* __restrict__ c,             // cell state [B,H] f32
    const float* __restrict__ bf_, const float* __restrict__ bi_,
    const float* __restrict__ bc_, const float* __restrict__ bo_) {
  extern __shared__ char smem[];
  __bf16* Alds = (__bf16*)smem;                                   // [mt][48][512]
  float*  Glds = (float*)(smem + (size_t)MT * KT_ALL * FRAG * 2); // [4][64][16]

  const int tid  = threadIdx.x;
  const int lane = tid & 31;
  const int w    = tid >> 5;      // wave id: staging mt, then gate id
  const int nt   = blockIdx.x;    // hidden N-tile 0..63

  // ---- stage A fragments (x part then h part) into LDS; wave w does mt=w
  {
    const int mt = w;
    const size_t lo = (size_t)lane * 16;
    for (int kt = 0; kt < KT_X; ++kt) {
      v16bf a = *(const v16bf*)(xp_s + (size_t)(mt * KT_X + kt) * FRAG + lo);
      *(v16bf*)(Alds + (size_t)(mt * KT_ALL + kt) * FRAG + lo) = a;
    }
    for (int kt = 0; kt < KT_H; ++kt) {
      v16bf a = *(const v16bf*)(hin + (size_t)(mt * KT_H + kt) * FRAG + lo);
      *(v16bf*)(Alds + (size_t)(mt * KT_ALL + KT_X + kt) * FRAG + lo) = a;
    }
  }
  __syncthreads();

  // ---- GEMM: wave w computes gate g = w for all 4 M-tiles of this N-tile.
  // Software-pipelined: loads for kt+1 issue before the 4 WMMAs of kt.
  const int g = w;
  const __bf16* bptr = wp + (size_t)((g * NT_H + nt) * KT_ALL) * FRAG + lane * 16;
  const __bf16* aptr = Alds + lane * 16;
  v8f acc0 = {}, acc1 = {}, acc2 = {}, acc3 = {};

  v16bf bc = *(const v16bf*)(bptr);
  v16bf a0 = *(const v16bf*)(aptr);
  v16bf a1 = *(const v16bf*)(aptr + 1 * KT_ALL * FRAG);
  v16bf a2 = *(const v16bf*)(aptr + 2 * KT_ALL * FRAG);
  v16bf a3 = *(const v16bf*)(aptr + 3 * KT_ALL * FRAG);
  for (int kt = 0; kt < KT_ALL - 1; ++kt) {
    v16bf bn = *(const v16bf*)(bptr + (size_t)(kt + 1) * FRAG);
    const __bf16* an = aptr + (size_t)(kt + 1) * FRAG;
    v16bf n0 = *(const v16bf*)(an);
    v16bf n1 = *(const v16bf*)(an + 1 * KT_ALL * FRAG);
    v16bf n2 = *(const v16bf*)(an + 2 * KT_ALL * FRAG);
    v16bf n3 = *(const v16bf*)(an + 3 * KT_ALL * FRAG);
    __builtin_prefetch(bptr + (size_t)(kt + 8) * FRAG, 0, 1);  // global_prefetch_b8
    acc0 = wmma_bf16(a0, bc, acc0);
    acc1 = wmma_bf16(a1, bc, acc1);
    acc2 = wmma_bf16(a2, bc, acc2);
    acc3 = wmma_bf16(a3, bc, acc3);
    bc = bn; a0 = n0; a1 = n1; a2 = n2; a3 = n3;
  }
  acc0 = wmma_bf16(a0, bc, acc0);
  acc1 = wmma_bf16(a1, bc, acc1);
  acc2 = wmma_bf16(a2, bc, acc2);
  acc3 = wmma_bf16(a3, bc, acc3);

  // ---- bias + activation -> gate LDS buffer
  const float* biasp = (g == 0) ? bf_ : (g == 1) ? bi_ : (g == 2) ? bc_ : bo_;
  const float bias   = biasp[nt * 16 + (lane & 15)];
  activate_store(Glds, acc0, g, 0, lane, bias);
  activate_store(Glds, acc1, g, 1, lane, bias);
  activate_store(Glds, acc2, g, 2, lane, bias);
  activate_store(Glds, acc3, g, 3, lane, bias);
  __syncthreads();

  // ---- fused elementwise: c = c*f + g*i ; h = c*o ; scatter h to A-frags
#pragma unroll
  for (int q = 0; q < 8; ++q) {
    int e  = tid * 8 + q;          // 0..1023 over [b=0..63][jj=0..15]
    int b  = e >> 4, jj = e & 15;
    float f  = Glds[0 * (BATCH * 16) + b * 16 + jj];
    float i  = Glds[1 * (BATCH * 16) + b * 16 + jj];
    float gg = Glds[2 * (BATCH * 16) + b * 16 + jj];
    float o  = Glds[3 * (BATCH * 16) + b * 16 + jj];
    int j = nt * 16 + jj;
    size_t cIdx = (size_t)b * HIDDEN + j;
    float cn = c[cIdx] * f + gg * i;
    float h  = cn * o;             // reference: no tanh on c
    c[cIdx] = cn;
    // inverse of A-layout: (b, j) -> fragment (mt, ktH), lane, slot
    int mt = b >> 4, m = b & 15;
    int ktH = j >> 5, kk = j & 31;
    int gA = (kk >> 3) & 1;
    int hi = (kk >> 4) & 1;
    int p  = kk & 7;
    int slot = (((hi ? 4 : 0) + (p >> 1)) << 1) | (kk & 1);
    int lt = m + 16 * gA;
    hout[(size_t)(mt * KT_H + ktH) * FRAG + lt * 16 + slot] = (__bf16)h;
  }
}

// ------------------------- final projection --------------------------------

// out[B,V] = h @ Wout + bout. grid = 250, block = 256 (8 waves, 1 N-tile each).
__global__ void __launch_bounds__(256) proj(
    const __bf16* __restrict__ hfin,   // final h A-frags: [mt][kt<32][512]
    const __bf16* __restrict__ wop,    // Wout B-frags: [nt][kt][512]
    const float* __restrict__ bout, float* __restrict__ out) {
  const int tid  = threadIdx.x;
  const int lane = tid & 31;
  const int w    = tid >> 5;
  const int nt   = blockIdx.x * 8 + w;           // 0..1999 exact
  const __bf16* bptr = wop + (size_t)(nt * KT_H) * FRAG + lane * 16;
  const __bf16* aptr = hfin + lane * 16;
  v8f acc0 = {}, acc1 = {}, acc2 = {}, acc3 = {};

  v16bf bc = *(const v16bf*)(bptr);
  v16bf a0 = *(const v16bf*)(aptr);
  v16bf a1 = *(const v16bf*)(aptr + 1 * KT_H * FRAG);
  v16bf a2 = *(const v16bf*)(aptr + 2 * KT_H * FRAG);
  v16bf a3 = *(const v16bf*)(aptr + 3 * KT_H * FRAG);
  for (int kt = 0; kt < KT_H - 1; ++kt) {
    v16bf bn = *(const v16bf*)(bptr + (size_t)(kt + 1) * FRAG);
    const __bf16* an = aptr + (size_t)(kt + 1) * FRAG;
    v16bf n0 = *(const v16bf*)(an);
    v16bf n1 = *(const v16bf*)(an + 1 * KT_H * FRAG);
    v16bf n2 = *(const v16bf*)(an + 2 * KT_H * FRAG);
    v16bf n3 = *(const v16bf*)(an + 3 * KT_H * FRAG);
    __builtin_prefetch(bptr + (size_t)(kt + 8) * FRAG, 0, 1);
    acc0 = wmma_bf16(a0, bc, acc0);
    acc1 = wmma_bf16(a1, bc, acc1);
    acc2 = wmma_bf16(a2, bc, acc2);
    acc3 = wmma_bf16(a3, bc, acc3);
    bc = bn; a0 = n0; a1 = n1; a2 = n2; a3 = n3;
  }
  acc0 = wmma_bf16(a0, bc, acc0);
  acc1 = wmma_bf16(a1, bc, acc1);
  acc2 = wmma_bf16(a2, bc, acc2);
  acc3 = wmma_bf16(a3, bc, acc3);

  const int col   = nt * 16 + (lane & 15);
  const float bias = bout[col];
  const int rbase = 8 * (lane >> 4);
  v8f accs[4] = {acc0, acc1, acc2, acc3};
#pragma unroll
  for (int mt = 0; mt < 4; ++mt) {
#pragma unroll
    for (int r = 0; r < 8; ++r) {
      int brow = mt * 16 + rbase + r;
      out[(size_t)brow * VOCAB + col] = accs[mt][r] + bias;
    }
  }
}

// ------------------------- host launcher -----------------------------------

extern "C" void kernel_launch(void* const* d_in, const int* in_sizes, int n_in,
                              void* d_out, int out_size, void* d_ws, size_t ws_size,
                              hipStream_t stream) {
  const int*   ids   = (const int*)d_in[0];
  const float* embed = (const float*)d_in[1];
  const float* Wf    = (const float*)d_in[2];
  const float* bf_   = (const float*)d_in[3];
  const float* Wi    = (const float*)d_in[4];
  const float* bi_   = (const float*)d_in[5];
  const float* Wc    = (const float*)d_in[6];
  const float* bc_   = (const float*)d_in[7];
  const float* Wo    = (const float*)d_in[8];
  const float* bo_   = (const float*)d_in[9];
  const float* Wout  = (const float*)d_in[10];
  const float* bout  = (const float*)d_in[11];
  float* out = (float*)d_out;

  // Workspace carve-up (bf16 regions first, then f32 cell state): ~107 MB.
  __bf16* Xp  = (__bf16*)d_ws;                                  // SEQ*MT*KT_X*FRAG
  __bf16* Wp  = Xp  + (size_t)SEQ * MT * KT_X * FRAG;           // 4*NT_H*KT_ALL*FRAG
  __bf16* Wop = Wp  + (size_t)4 * NT_H * KT_ALL * FRAG;         // NT_V*KT_H*FRAG
  __bf16* Hp  = Wop + (size_t)NT_V * KT_H * FRAG;               // 2 * (BATCH*HIDDEN)
  float*  cbuf = (float*)(Hp + (size_t)2 * BATCH * HIDDEN);     // BATCH*HIDDEN

  const size_t HBUF = (size_t)BATCH * HIDDEN;                   // 65536 elems
  const size_t XSTEP = (size_t)MT * KT_X * FRAG;                // 32768 elems/step
  const int dynLds = MT * KT_ALL * FRAG * 2 /*A bf16*/ + 4 * BATCH * 16 * 4 /*gates f32*/;

  // Allow >64KB dynamic LDS (gfx1250 WGP supports 320KB).
  (void)hipFuncSetAttribute((const void*)lstm_step,
                            hipFuncAttributeMaxDynamicSharedMemorySize, dynLds);

  zero_state<<<256, 256, 0, stream>>>(cbuf, Hp);                // h0 := Hp[0], c := 0
  pack_x   <<<4096, 256, 0, stream>>>(ids, embed, Xp);
  pack_w   <<<1536, 256, 0, stream>>>(Wf, Wi, Wc, Wo, Wp);
  pack_wout<<<8000, 256, 0, stream>>>(Wout, Wop);

  for (int s = 0; s < SEQ; ++s) {
    const __bf16* hin  = Hp + (size_t)(s & 1) * HBUF;
    __bf16*       hout = Hp + (size_t)((s + 1) & 1) * HBUF;
    lstm_step<<<NT_H, 128, dynLds, stream>>>(Xp + (size_t)s * XSTEP, Wp, hin, hout,
                                             cbuf, bf_, bi_, bc_, bo_);
  }
  // SEQ even -> final h lives in buffer 0.
  proj<<<NT_V / 8, 256, 0, stream>>>(Hp, Wop, bout, out);
}
